// TransformerBlock_15968688406763
// MI455X (gfx1250) — matrix operand order repaired and checked by reading, not compiled, not run on hardware
//
#include <hip/hip_runtime.h>
#include <hip/hip_bf16.h>

#define B_  2
#define S_  2048
#define E_  1024
#define H_  16
#define D_  64
#define FF_ 4096

typedef __attribute__((ext_vector_type(16))) __bf16 v16bf;
typedef __attribute__((ext_vector_type(8)))  __bf16 v8bf;
typedef __attribute__((ext_vector_type(8)))  float  v8f;

#define DEV static __device__ __forceinline__

DEV v8f wmma_bf16(v16bf a, v16bf b, v8f c) {
  // D = A(16x32 bf16) x B(32x16 bf16) + C(16x16 f32)
  return __builtin_amdgcn_wmma_f32_16x16x32_bf16(
      /*neg_a=*/false, a, /*neg_b=*/false, b,
      /*c_mod=*/(short)0, c, /*reuse_a=*/false, /*reuse_b=*/false);
}

DEV v16bf cat8(v8bf lo, v8bf hi) {
  return __builtin_shufflevector(lo, hi, 0,1,2,3,4,5,6,7,8,9,10,11,12,13,14,15);
}
// A-fragment: per-lane pointer already at (row, kbase); needs elems [0..7] and [16..23]
DEV v16bf ldA(const __bf16* p) {
  return cat8(*(const v8bf*)p, *(const v8bf*)(p + 16));
}
// B-fragment: per-lane pointer at (col, kbase); needs elems [0..7] and [16..23] (B staged transposed in LDS)
DEV v16bf ldB(const __bf16* p) {
  return cat8(*(const v8bf*)p, *(const v8bf*)(p + 16));
}

// ---------------------------------------------------------------------------
// Kernel 1: per-head QKV projection. X[B*S,E] (f32) x W[64,64] -> bf16 [B,H,S,D]
// grid = (B*S/16, H/4), block = 128 (4 waves, one head each)
// ---------------------------------------------------------------------------
__global__ __launch_bounds__(128)
void qkv_kernel(const float* __restrict__ xq, const float* __restrict__ xk,
                const float* __restrict__ xv,
                const float* __restrict__ wq, const float* __restrict__ wk,
                const float* __restrict__ wv,
                __bf16* __restrict__ qo, __bf16* __restrict__ ko,
                __bf16* __restrict__ vo) {
  __shared__ __align__(16) __bf16 sWT[3][64][72];  // [mat][n][k] (transposed weights)
  __shared__ __align__(16) __bf16 sX[3][16][256];  // [mat][row][4-head col slice]
  const int tid = threadIdx.x;
  const int lane = tid & 31, wid = tid >> 5;
  const int r0 = blockIdx.x * 16;   // flat row in [0, B*S)
  const int c0 = blockIdx.y * 256;  // column base of 4-head group
  const float* Xs[3] = {xq, xk, xv};
  const float* Ws[3] = {wq, wk, wv};
  __bf16*      Os[3] = {qo, ko, vo};

  for (int i = tid; i < 3 * 64 * 64; i += 128) {
    int mat = i >> 12, rem = i & 4095;
    int kk = rem >> 6, nn = rem & 63;
    sWT[mat][nn][kk] = (__bf16)Ws[mat][kk * 64 + nn];
  }
  for (int i = tid; i < 3 * 16 * 256; i += 128) {
    int mat = i >> 12, rem = i & 4095;
    int rr = rem >> 8, cc = rem & 255;
    sX[mat][rr][cc] = (__bf16)Xs[mat][(size_t)(r0 + rr) * E_ + c0 + cc];
  }
  __syncthreads();

  const int h = blockIdx.y * 4 + wid;
  const int woff = wid * 64;
  const int m = lane & 15;
  const int kbaseA = (lane >> 4) << 3;  // 0 or 8
  const int kbaseB = (lane >> 4) << 3;  // 0 or 8
  const int mb = (lane >> 4) << 3;

  v8f zero = {};
#pragma unroll
  for (int mat = 0; mat < 3; ++mat) {
    v16bf a0 = ldA(&sX[mat][m][woff + 0 + kbaseA]);
    v16bf a1 = ldA(&sX[mat][m][woff + 32 + kbaseA]);
    __bf16* op = Os[mat];
#pragma unroll
    for (int nt = 0; nt < 4; ++nt) {
      v8f acc = zero;
      acc = wmma_bf16(a0, ldB(&sWT[mat][nt * 16 + m][0 + kbaseB]), acc);
      acc = wmma_bf16(a1, ldB(&sWT[mat][nt * 16 + m][32 + kbaseB]), acc);
#pragma unroll
      for (int r = 0; r < 8; ++r) {
        int row = r0 + mb + r;
        int bb = row >> 11;         // / S_
        int ss = row & (S_ - 1);
        op[(((size_t)bb * H_ + h) * S_ + ss) * D_ + nt * 16 + m] = (__bf16)acc[r];
      }
    }
  }
}

// ---------------------------------------------------------------------------
// Kernel 2: flash attention. grid = (B*H, S/64), block = 128 (4 q-tiles share K/V)
// ---------------------------------------------------------------------------
__global__ __launch_bounds__(128)
void attn_kernel(const __bf16* __restrict__ q_bf, const __bf16* __restrict__ k_bf,
                 const __bf16* __restrict__ v_bf, const int* __restrict__ mask,
                 float* __restrict__ attn_out) {
  __shared__ __align__(16) __bf16 sK[32][72];   // [key][d]   (row-major: B-frag for Q@K^T)
  __shared__ __align__(16) __bf16 sVT[64][40];  // [d][key]   (transposed: B-frag for P@V)
  __shared__ __align__(16) __bf16 sP[4][16][32];
  const int tid = threadIdx.x, lane = tid & 31, wid = tid >> 5;
  const int bh = blockIdx.x, b = bh >> 4;
  const int h = bh & (H_ - 1);
  const int q0 = blockIdx.y * 64 + wid * 16;
  const __bf16* qb  = q_bf + (size_t)bh * S_ * D_;
  const __bf16* kbp = k_bf + (size_t)bh * S_ * D_;
  const __bf16* vbp = v_bf + (size_t)bh * S_ * D_;
  const int m = lane & 15;
  const int kbaseA = (lane >> 4) << 3;
  const int kbaseB = (lane >> 4) << 3;
  const int mb = (lane >> 4) << 3;

  v16bf aq0 = ldA(qb + (size_t)(q0 + m) * D_ + 0 + kbaseA);
  v16bf aq1 = ldA(qb + (size_t)(q0 + m) * D_ + 32 + kbaseA);

  v8f zero = {};
  v8f o0 = zero, o1 = zero, o2 = zero, o3 = zero;
  float mi[8], li[8];
#pragma unroll
  for (int r = 0; r < 8; ++r) { mi[r] = -3.0e38f; li[r] = 0.f; }

  for (int ck = 0; ck < S_ / 32; ++ck) {
    __syncthreads();
    for (int i = tid; i < 32 * 64; i += 128) {
      int rr = i >> 6, cc = i & 63;
      sK[rr][cc]  = kbp[(size_t)(ck * 32 + rr) * D_ + cc];
      sVT[cc][rr] = vbp[(size_t)(ck * 32 + rr) * D_ + cc];
    }
    if (ck + 1 < S_ / 32) {
      __builtin_prefetch(kbp + (size_t)(ck + 1) * 32 * D_, 0, 3);
      __builtin_prefetch(vbp + (size_t)(ck + 1) * 32 * D_, 0, 3);
    }
    __syncthreads();

    // scores 16x32 for this wave's q-tile: two 16x16 D tiles, K=64 in two chunks
    v8f s0 = zero, s1 = zero;
    s0 = wmma_bf16(aq0, ldB(&sK[0  + m][0  + kbaseB]), s0);
    s0 = wmma_bf16(aq1, ldB(&sK[0  + m][32 + kbaseB]), s0);
    s1 = wmma_bf16(aq0, ldB(&sK[16 + m][0  + kbaseB]), s1);
    s1 = wmma_bf16(aq1, ldB(&sK[16 + m][32 + kbaseB]), s1);

    const float scl = 0.03125f;  // 1/sqrt(EMBED) = 1/32
    const int key0 = ck * 32 + m;
    const float bias0 = mask[b * S_ + key0]      ? 0.f : -1.0e20f;
    const float bias1 = mask[b * S_ + key0 + 16] ? 0.f : -1.0e20f;
#pragma unroll
    for (int r = 0; r < 8; ++r) {
      s0[r] = s0[r] * scl + bias0;
      s1[r] = s1[r] * scl + bias1;
    }

    // online softmax per row (rows live in 16-lane groups)
#pragma unroll
    for (int r = 0; r < 8; ++r) {
      float t = fmaxf(s0[r], s1[r]);
      t = fmaxf(t, __shfl_xor(t, 1, 32));
      t = fmaxf(t, __shfl_xor(t, 2, 32));
      t = fmaxf(t, __shfl_xor(t, 4, 32));
      t = fmaxf(t, __shfl_xor(t, 8, 32));
      float nm = fmaxf(mi[r], t);
      float corr = __expf(mi[r] - nm);
      mi[r] = nm;
      li[r] *= corr;
      o0[r] *= corr; o1[r] *= corr; o2[r] *= corr; o3[r] *= corr;
      float p0 = __expf(s0[r] - nm);
      float p1 = __expf(s1[r] - nm);
      sP[wid][mb + r][m]      = (__bf16)p0;
      sP[wid][mb + r][16 + m] = (__bf16)p1;
      float ls = p0 + p1;
      ls += __shfl_xor(ls, 1, 32);
      ls += __shfl_xor(ls, 2, 32);
      ls += __shfl_xor(ls, 4, 32);
      ls += __shfl_xor(ls, 8, 32);
      li[r] += ls;
    }
    asm volatile("s_wait_dscnt 0x0" ::: "memory");  // per-wave P handoff via LDS

    // O += P(16x32) @ V(32x64)
    v16bf ap = ldA(&sP[wid][m][kbaseA]);
    o0 = wmma_bf16(ap, ldB(&sVT[0  + m][kbaseB]), o0);
    o1 = wmma_bf16(ap, ldB(&sVT[16 + m][kbaseB]), o1);
    o2 = wmma_bf16(ap, ldB(&sVT[32 + m][kbaseB]), o2);
    o3 = wmma_bf16(ap, ldB(&sVT[48 + m][kbaseB]), o3);
  }

#pragma unroll
  for (int r = 0; r < 8; ++r) {
    float inv = 1.f / li[r];
    int srow = q0 + mb + r;
    size_t base = ((size_t)b * S_ + srow) * E_ + h * D_ + m;
    attn_out[base]      = o0[r] * inv;
    attn_out[base + 16] = o1[r] * inv;
    attn_out[base + 32] = o2[r] * inv;
    attn_out[base + 48] = o3[r] * inv;
  }
}

// ---------------------------------------------------------------------------
// Kernel 3: generic WMMA GEMM  C = epi(A @ Bw + bias). 64x64 tile, 4 waves.
// grid = (M/64, N/64), block = 128
// ---------------------------------------------------------------------------
template <int RELU, typename AT, typename CT>
__global__ __launch_bounds__(128)
void gemm_kernel(const AT* __restrict__ A, const float* __restrict__ Bw,
                 const float* __restrict__ bias, CT* __restrict__ C,
                 int M, int N, int K) {
  __shared__ __align__(16) __bf16 sA[64][40];   // [m][k]
  __shared__ __align__(16) __bf16 sBT[64][40];  // [n][k] (transposed weight chunk)
  const int tid = threadIdx.x, lane = tid & 31, wid = tid >> 5;
  const int m0 = blockIdx.x * 64, n0 = blockIdx.y * 64;
  const int m = lane & 15, mb = (lane >> 4) << 3;
  const int kbaseA = (lane >> 4) << 3, kbaseB = (lane >> 4) << 3;
  v8f zero = {};
  v8f acc0 = zero, acc1 = zero, acc2 = zero, acc3 = zero;

  for (int k0 = 0; k0 < K; k0 += 32) {
    __syncthreads();
    for (int i = tid; i < 2048; i += 128) {
      int ra = i >> 5, ca = i & 31;
      sA[ra][ca] = (__bf16)(float)A[(size_t)(m0 + ra) * K + k0 + ca];
      int rb = i >> 6, cb = i & 63;
      sBT[cb][rb] = (__bf16)Bw[(size_t)(k0 + rb) * N + n0 + cb];
    }
    __syncthreads();
    v16bf a = ldA(&sA[wid * 16 + m][kbaseA]);
    acc0 = wmma_bf16(a, ldB(&sBT[0  + m][kbaseB]), acc0);
    acc1 = wmma_bf16(a, ldB(&sBT[16 + m][kbaseB]), acc1);
    acc2 = wmma_bf16(a, ldB(&sBT[32 + m][kbaseB]), acc2);
    acc3 = wmma_bf16(a, ldB(&sBT[48 + m][kbaseB]), acc3);
  }

#pragma unroll
  for (int r = 0; r < 8; ++r) {
    int mr = m0 + wid * 16 + mb + r;
    int nc = n0 + m;
    float v0 = acc0[r] + bias[nc];
    float v1 = acc1[r] + bias[nc + 16];
    float v2 = acc2[r] + bias[nc + 32];
    float v3 = acc3[r] + bias[nc + 48];
    if (RELU) {
      v0 = fmaxf(v0, 0.f); v1 = fmaxf(v1, 0.f);
      v2 = fmaxf(v2, 0.f); v3 = fmaxf(v3, 0.f);
    }
    size_t base = (size_t)mr * N + nc;
    C[base]      = (CT)v0;
    C[base + 16] = (CT)v1;
    C[base + 32] = (CT)v2;
    C[base + 48] = (CT)v3;
  }
}

// ---------------------------------------------------------------------------
// Kernel 4: out = LayerNorm(a + b) * g + beta  (row = 1024 cols). block = 256
// ---------------------------------------------------------------------------
DEV float block_sum(float v, float* red) {
#pragma unroll
  for (int o = 16; o >= 1; o >>= 1) v += __shfl_xor(v, o, 32);
  if ((threadIdx.x & 31) == 0) red[threadIdx.x >> 5] = v;
  __syncthreads();
  float t = 0.f;
#pragma unroll
  for (int i = 0; i < 8; ++i) t += red[i];
  __syncthreads();
  return t;
}

__global__ __launch_bounds__(256)
void add_ln_kernel(const float* __restrict__ a, const float* __restrict__ b,
                   const float* __restrict__ g, const float* __restrict__ bet,
                   float* __restrict__ out) {
  __shared__ float red[8];
  const int row = blockIdx.x;
  const float* ar = a + (size_t)row * E_;
  const float* br = b + (size_t)row * E_;
  float x[4];
  float s = 0.f;
#pragma unroll
  for (int i = 0; i < 4; ++i) {
    int c = (threadIdx.x << 2) + i;
    x[i] = ar[c] + br[c];
    s += x[i];
  }
  float mu = block_sum(s, red) * (1.f / E_);
  float vs = 0.f;
#pragma unroll
  for (int i = 0; i < 4; ++i) {
    float d = x[i] - mu;
    vs += d * d;
  }
  float rstd = rsqrtf(block_sum(vs, red) * (1.f / E_) + 1e-5f);
  float* orow = out + (size_t)row * E_;
#pragma unroll
  for (int i = 0; i < 4; ++i) {
    int c = (threadIdx.x << 2) + i;
    orow[c] = (x[i] - mu) * rstd * g[c] + bet[c];
  }
}

// ---------------------------------------------------------------------------
extern "C" void kernel_launch(void* const* d_in, const int* in_sizes, int n_in,
                              void* d_out, int out_size, void* d_ws, size_t ws_size,
                              hipStream_t stream) {
  (void)in_sizes; (void)n_in; (void)out_size; (void)ws_size;
  const float* value = (const float*)d_in[0];
  const float* key_  = (const float*)d_in[1];
  const float* query = (const float*)d_in[2];
  const int*   mask  = (const int*)d_in[3];
  const float* wv    = (const float*)d_in[4];
  const float* wk    = (const float*)d_in[5];
  const float* wq    = (const float*)d_in[6];
  const float* w_out = (const float*)d_in[7];
  const float* b_out = (const float*)d_in[8];
  const float* g1    = (const float*)d_in[9];
  const float* beta1 = (const float*)d_in[10];
  const float* w_ff1 = (const float*)d_in[11];
  const float* b_ff1 = (const float*)d_in[12];
  const float* w_ff2 = (const float*)d_in[13];
  const float* b_ff2 = (const float*)d_in[14];
  const float* g2    = (const float*)d_in[15];
  const float* beta2 = (const float*)d_in[16];
  float* out = (float*)d_out;

  char* ws = (char*)d_ws;
  const size_t BS   = (size_t)B_ * S_;
  const size_t QKVB = BS * E_ * sizeof(__bf16);  // one [B,H,S,D] bf16 tensor
  const size_t F32B = BS * E_ * sizeof(float);
  __bf16* q_bf   = (__bf16*)ws; ws += QKVB;
  __bf16* k_bf   = (__bf16*)ws; ws += QKVB;
  __bf16* v_bf   = (__bf16*)ws; ws += QKVB;
  float*  attn_o = (float*)ws;  ws += F32B;
  float*  tmp    = (float*)ws;  ws += F32B;   // proj result, later ff2 result
  float*  x1     = (float*)ws;  ws += F32B;
  __bf16* ff1_bf = (__bf16*)ws; ws += BS * FF_ * sizeof(__bf16);

  qkv_kernel<<<dim3(BS / 16, H_ / 4), 128, 0, stream>>>(
      query, key_, value, wq, wk, wv, q_bf, k_bf, v_bf);

  attn_kernel<<<dim3(B_ * H_, S_ / 64), 128, 0, stream>>>(
      q_bf, k_bf, v_bf, mask, attn_o);

  gemm_kernel<0, float, float><<<dim3(BS / 64, E_ / 64), 128, 0, stream>>>(
      attn_o, w_out, b_out, tmp, (int)BS, E_, E_);

  add_ln_kernel<<<dim3(BS), 256, 0, stream>>>(tmp, query, g1, beta1, x1);

  gemm_kernel<1, float, __bf16><<<dim3(BS / 64, FF_ / 64), 128, 0, stream>>>(
      x1, w_ff1, b_ff1, ff1_bf, (int)BS, FF_, E_);

  gemm_kernel<0, __bf16, float><<<dim3(BS / 64, E_ / 64), 128, 0, stream>>>(
      ff1_bf, w_ff2, b_ff2, tmp, (int)BS, E_, FF_);

  add_ln_kernel<<<dim3(BS), 256, 0, stream>>>(tmp, x1, g2, beta2, out);
}
